// LSTM_regression_1228360647076
// MI455X (gfx1250) — matrix operand order repaired and checked by reading, not compiled, not run on hardware
//
#include <hip/hip_runtime.h>

// Problem dims (from reference)
#define S_LEN  2048
#define BATCH  256
#define IN_DIM 64
#define HID    256
#define FOURH  1024

#define NBLK  16   // persistent workgroups; each owns 16 h-columns for both layers
#define TPB   512  // 16 wave32 waves; each wave owns one 16-row M tile
#define BH    (BATCH * HID)

// LDS weight slice layout (ushort elements), padded to avoid bank conflicts:
// row stride K+8 => consecutive N rows land 4 banks apart for K=256 and K=64.
#define WX0_STRIDE (IN_DIM + 8)                 // 72
#define WH_STRIDE  (HID + 8)                    // 264
#define WX0_ELEMS  (64 * WX0_STRIDE)            // 4608
#define WH_ELEMS   (64 * WH_STRIDE)             // 16896
#define SMEM_ELEMS (WX0_ELEMS + 3 * WH_ELEMS)   // 55296
#define SMEM_BYTES (SMEM_ELEMS * 2)             // 110592 (< 320KB WGP LDS)

typedef __attribute__((ext_vector_type(16))) __bf16 v16bf;
typedef __attribute__((ext_vector_type(8)))  float  v8f;

// ---------------- helpers ----------------

__device__ __forceinline__ unsigned short f2bf(float f) {
  unsigned x = __builtin_bit_cast(unsigned, f);
  unsigned r = x + 0x7FFFu + ((x >> 16) & 1u);   // round-to-nearest-even
  return (unsigned short)(r >> 16);
}

__device__ __forceinline__ float sigmoidf_(float z) {
  return 1.0f / (1.0f + __expf(-z));
}

__device__ __forceinline__ v16bf frag_cast(uint4 lo, uint4 hi) {
  union { uint4 q[2]; v16bf v; } u;
  u.q[0] = lo; u.q[1] = hi;
  return u.v;
}

// A-matrix fragment (16x32 bf16) from row-major [rows][ld] bf16 memory.
// CDNA5 ISA 7.12.2: lanes 0-15 -> M=lane, K chunks {0..7,16..23};
// lanes 16-31 -> M=lane-16, K chunks {8..15,24..31}.
__device__ __forceinline__ v16bf load_a_bf(const unsigned short* __restrict__ base,
                                           int ld, int mbase, int kb, int lane) {
  const unsigned short* p =
      base + (size_t)(mbase + (lane & 15)) * ld + kb + ((lane >> 4) << 3);
  return frag_cast(*(const uint4*)p, *(const uint4*)(p + 16));
}

// B-matrix fragment (32x16 bf16) from LDS [64 rows][stride] (row = g*16+n, K contiguous).
// Lanes 0-15 -> N=lane, K chunks {0..7,16..23}; lanes 16-31 -> N=lane-16, K chunks {8..15,24..31}.
__device__ __forceinline__ v16bf load_b_lds(const unsigned short* sW, int stride,
                                            int g, int kb, int lane) {
  const unsigned short* p =
      sW + (size_t)((g << 4) + (lane & 15)) * stride + kb + ((lane >> 4) << 3);
  return frag_cast(*(const uint4*)p, *(const uint4*)(p + 16));
}

__device__ __forceinline__ v8f wmma_bf16(v16bf a, v16bf b, v8f c) {
  return __builtin_amdgcn_wmma_f32_16x16x32_bf16(
      /*neg_a=*/false, a, /*neg_b=*/false, b,
      /*c_mod=*/(short)0, c, /*reuse_a=*/false, /*reuse_b=*/false);
}

// Release/acquire software grid barrier across NBLK persistent workgroups.
// The leading s_cluster_barrier is the HW fast path when the 16-WG dispatch is
// cluster-formed (ISA: S_NOP when ClusterID==0, so it is safe either way).
__device__ __forceinline__ void grid_barrier(unsigned* bar, unsigned target) {
  __syncthreads();
  __builtin_amdgcn_s_cluster_barrier();
  if (threadIdx.x == 0) {
    __threadfence();
    if (atomicAdd(&bar[0], 1u) == NBLK - 1) {
      bar[0] = 0;
      __threadfence();
      atomicExch(&bar[1], target);
    } else {
      while (atomicAdd(&bar[1], 0u) < target) __builtin_amdgcn_s_sleep(2);
    }
    __threadfence();
  }
  __syncthreads();
}

// ---------------- setup kernels ----------------

__global__ void cvt_bf16(const float* __restrict__ src,
                         unsigned short* __restrict__ dst, int n) {
  int i = blockIdx.x * blockDim.x + threadIdx.x;
  if (i < n) dst[i] = f2bf(src[i]);
}

__global__ void bias_add(const float* __restrict__ a, const float* __restrict__ b,
                         float* __restrict__ dst, int n) {
  int i = blockIdx.x * blockDim.x + threadIdx.x;
  if (i < n) dst[i] = a[i] + b[i];
}

__global__ void init_state(float* __restrict__ out, int nOut,
                           const float* __restrict__ bfc,
                           unsigned short* __restrict__ H1,
                           unsigned short* __restrict__ H2,
                           unsigned* __restrict__ bar) {
  int i = blockIdx.x * blockDim.x + threadIdx.x;
  int stride = gridDim.x * blockDim.x;
  float bias = bfc[0];
  for (int j = i; j < nOut; j += stride) out[j] = bias;       // FC bias pre-seeded
  for (int j = i; j < 2 * BH; j += stride) { H1[j] = 0; H2[j] = 0; }
  if (i == 0) { bar[0] = 0; bar[1] = 0; }
}

// ---------------- persistent fused 2-layer LSTM + FC ----------------

__global__ __launch_bounds__(TPB, 1)
void lstm_fused(const unsigned short* __restrict__ Xbf,
                const unsigned short* __restrict__ Wx0,
                const unsigned short* __restrict__ Wh0,
                const float* __restrict__ bias0,
                const unsigned short* __restrict__ Wx1,
                const unsigned short* __restrict__ Wh1,
                const float* __restrict__ bias1,
                const float* __restrict__ wfc,
                unsigned short* __restrict__ H1buf,
                unsigned short* __restrict__ H2buf,
                unsigned* __restrict__ bar,
                float* __restrict__ out) {
  extern __shared__ unsigned short smem[];
  unsigned short* sWx0 = smem;                     // [64][WX0_STRIDE]
  unsigned short* sWh0 = smem + WX0_ELEMS;         // [64][WH_STRIDE]
  unsigned short* sWx1 = sWh0 + WH_ELEMS;
  unsigned short* sWh1 = sWx1 + WH_ELEMS;

  const int wg   = blockIdx.x;             // owns h columns [wg*16, wg*16+16)
  const int tid  = threadIdx.x;
  const int lane = tid & 31;
  const int wave = tid >> 5;               // 0..15
  const int mb   = wave << 4;              // this wave's batch rows [mb, mb+16)
  const int hcol = (wg << 4) + (lane & 15);
  const int mOff = (lane >> 4) << 3;       // D-matrix row offset within a tile
  const int nb   = wg << 4;

  // ---- stage this WG's weight slices into LDS (reused for 2048 steps) ----
  {
    // rows are (g*16 + n) -> global row g*HID + nb + n ; copy uint4 chunks
    auto stage = [&](const unsigned short* __restrict__ gsrc, int K,
                     unsigned short* sdst, int sstride) {
      int chunksPerRow = K >> 3;
      for (int idx = tid; idx < 64 * chunksPerRow; idx += TPB) {
        int r = idx / chunksPerRow;
        int c = idx - r * chunksPerRow;
        int grow = ((r >> 4) * HID) + nb + (r & 15);
        *(uint4*)(sdst + r * sstride + (c << 3)) =
            *(const uint4*)(gsrc + (size_t)grow * K + (c << 3));
      }
    };
    stage(Wx0, IN_DIM, sWx0, WX0_STRIDE);
    stage(Wh0, HID,    sWh0, WH_STRIDE);
    stage(Wx1, HID,    sWx1, WH_STRIDE);
    stage(Wh1, HID,    sWh1, WH_STRIDE);
  }
  __syncthreads();

  float c1[8], c2[8];
#pragma unroll
  for (int i = 0; i < 8; ++i) { c1[i] = 0.0f; c2[i] = 0.0f; }

  v8f zero;
#pragma unroll
  for (int k = 0; k < 8; ++k) zero[k] = 0.0f;

  const float b0i = bias0[0 * HID + hcol], b0f = bias0[1 * HID + hcol];
  const float b0g = bias0[2 * HID + hcol], b0o = bias0[3 * HID + hcol];
  const float b1i = bias1[0 * HID + hcol], b1f = bias1[1 * HID + hcol];
  const float b1g = bias1[2 * HID + hcol], b1o = bias1[3 * HID + hcol];
  const float wf  = wfc[hcol];

  // K-loop over one A operand against 4 gate-column tiles held in LDS.
  v8f acc[4];
  auto gemm_into = [&](const unsigned short* __restrict__ A, int ldA, int K,
                       const unsigned short* sW, int sstride) {
    for (int kb = 0; kb < K; kb += 32) {
      v16bf a = load_a_bf(A, ldA, mb, kb, lane);
#pragma unroll
      for (int g = 0; g < 4; ++g) {
        v16bf bW = load_b_lds(sW, sstride, g, kb, lane);
        acc[g] = wmma_bf16(a, bW, acc[g]);
      }
    }
  };

  unsigned barGen = 0;

  // Software pipeline with 1-step skew: phase t = { layer1(t-1); layer0(t) }
  // followed by a single grid barrier (2049 barriers total instead of 4096).
  for (int t = 0; t <= S_LEN; ++t) {
    // ---------------- layer 1 of step (t-1) + FC head -------------------
    if (t > 0) {
      const int tp = t - 1;
      const unsigned short* h1in  = H1buf + (size_t)(tp & 1) * BH;         // h1(tp)
      const unsigned short* h2in  = H2buf + (size_t)((tp & 1) ^ 1) * BH;   // h2(tp-1)
      unsigned short*       h2out = H2buf + (size_t)(tp & 1) * BH;         // h2(tp)

#pragma unroll
      for (int g = 0; g < 4; ++g) acc[g] = zero;
      gemm_into(h1in, HID, HID, sWx1, WH_STRIDE);
      gemm_into(h2in, HID, HID, sWh1, WH_STRIDE);

#pragma unroll
      for (int r = 0; r < 8; ++r) {
        int m = mb + mOff + r;
        float gi = sigmoidf_(acc[0][r] + b1i);
        float gf = sigmoidf_(acc[1][r] + b1f);
        float gg = tanhf(acc[2][r] + b1g);
        float go = sigmoidf_(acc[3][r] + b1o);
        float c  = gf * c2[r] + gi * gg;
        c2[r] = c;
        float h  = go * tanhf(c);
        h2out[(size_t)m * HID + hcol] = f2bf(h);

        // FC: partial dot over this WG's 16 h-columns, reduced across the 16
        // lanes sharing batch row m, accumulated into out[tp*B + m].
        float part = h * wf;
#pragma unroll
        for (int sft = 1; sft < 16; sft <<= 1) part += __shfl_xor(part, sft, 32);
        if ((lane & 15) == 0) atomicAdd(&out[(size_t)tp * BATCH + m], part);
      }
    }

    // ---------------- layer 0 of step t ---------------------------------
    if (t < S_LEN) {
      const unsigned short* xin   = Xbf + (size_t)t * BATCH * IN_DIM;
      const unsigned short* h1in  = H1buf + (size_t)((t & 1) ^ 1) * BH;    // h1(t-1)
      unsigned short*       h1out = H1buf + (size_t)(t & 1) * BH;          // h1(t)

      if (t + 1 < S_LEN)  // keep next x tile warm in L2 (global_prefetch_b8)
        __builtin_prefetch(Xbf + (size_t)(t + 1) * BATCH * IN_DIM + (tid << 4), 0, 0);

#pragma unroll
      for (int g = 0; g < 4; ++g) acc[g] = zero;
      gemm_into(xin,  IN_DIM, IN_DIM, sWx0, WX0_STRIDE);
      gemm_into(h1in, HID,    HID,    sWh0, WH_STRIDE);

#pragma unroll
      for (int r = 0; r < 8; ++r) {
        int m = mb + mOff + r;
        float gi = sigmoidf_(acc[0][r] + b0i);
        float gf = sigmoidf_(acc[1][r] + b0f);
        float gg = tanhf(acc[2][r] + b0g);
        float go = sigmoidf_(acc[3][r] + b0o);
        float c  = gf * c1[r] + gi * gg;
        c1[r] = c;
        float h  = go * tanhf(c);
        h1out[(size_t)m * HID + hcol] = f2bf(h);
      }
    }

    __threadfence();
    grid_barrier(bar, ++barGen);
  }
}

// ---------------- host launcher ----------------

extern "C" void kernel_launch(void* const* d_in, const int* in_sizes, int n_in,
                              void* d_out, int out_size, void* d_ws, size_t ws_size,
                              hipStream_t stream) {
  const float* x     = (const float*)d_in[0];
  const float* w_ih0 = (const float*)d_in[1];
  const float* w_hh0 = (const float*)d_in[2];
  const float* b_ih0 = (const float*)d_in[3];
  const float* b_hh0 = (const float*)d_in[4];
  const float* w_ih1 = (const float*)d_in[5];
  const float* w_hh1 = (const float*)d_in[6];
  const float* b_ih1 = (const float*)d_in[7];
  const float* b_hh1 = (const float*)d_in[8];
  const float* w_fc  = (const float*)d_in[9];
  const float* b_fc  = (const float*)d_in[10];
  float* out = (float*)d_out;

  char* ws = (char*)d_ws;
  size_t off = 0;
  auto alloc = [&](size_t bytes) -> char* {
    off = (off + 255) & ~(size_t)255;
    char* p = ws + off;
    off += bytes;
    return p;
  };

  unsigned*       bar = (unsigned*)alloc(256);
  unsigned short* H1  = (unsigned short*)alloc((size_t)2 * BH * 2);
  unsigned short* H2  = (unsigned short*)alloc((size_t)2 * BH * 2);
  unsigned short* Wx0 = (unsigned short*)alloc((size_t)FOURH * IN_DIM * 2);
  unsigned short* Wh0 = (unsigned short*)alloc((size_t)FOURH * HID * 2);
  unsigned short* Wx1 = (unsigned short*)alloc((size_t)FOURH * HID * 2);
  unsigned short* Wh1 = (unsigned short*)alloc((size_t)FOURH * HID * 2);
  float*          bias0 = (float*)alloc((size_t)FOURH * 4);
  float*          bias1 = (float*)alloc((size_t)FOURH * 4);
  unsigned short* Xbf = (unsigned short*)alloc((size_t)S_LEN * BATCH * IN_DIM * 2);
  (void)ws_size; (void)in_sizes; (void)n_in;

  // One-time (per call) prep: narrow weights and the whole input to bf16.
  // fp32 [4H][K] row-major is already the [N][K] layout the B-fragment wants.
  const int nX = S_LEN * BATCH * IN_DIM;
  cvt_bf16<<<(nX + 255) / 256, 256, 0, stream>>>(x, Xbf, nX);
  cvt_bf16<<<(FOURH * IN_DIM + 255) / 256, 256, 0, stream>>>(w_ih0, Wx0, FOURH * IN_DIM);
  cvt_bf16<<<(FOURH * HID + 255) / 256, 256, 0, stream>>>(w_hh0, Wh0, FOURH * HID);
  cvt_bf16<<<(FOURH * HID + 255) / 256, 256, 0, stream>>>(w_ih1, Wx1, FOURH * HID);
  cvt_bf16<<<(FOURH * HID + 255) / 256, 256, 0, stream>>>(w_hh1, Wh1, FOURH * HID);
  bias_add<<<(FOURH + 255) / 256, 256, 0, stream>>>(b_ih0, b_hh0, bias0, FOURH);
  bias_add<<<(FOURH + 255) / 256, 256, 0, stream>>>(b_ih1, b_hh1, bias1, FOURH);
  init_state<<<64, 256, 0, stream>>>(out, out_size, b_fc, H1, H2, bar);

  // Allow >64KB dynamic LDS (WGP has 320KB); harmless if already permitted.
  (void)hipFuncSetAttribute((const void*)lstm_fused,
                            hipFuncAttributeMaxDynamicSharedMemorySize,
                            (int)SMEM_BYTES);

  lstm_fused<<<NBLK, TPB, SMEM_BYTES, stream>>>(Xbf, Wx0, Wh0, bias0,
                                                Wx1, Wh1, bias1, w_fc,
                                                H1, H2, bar, out);
}
